// LoRAAttention_25520695672940
// MI455X (gfx1250) — matrix operand order repaired and checked by reading, compiled, not run on hardware
//
#include <hip/hip_runtime.h>

typedef __attribute__((ext_vector_type(16))) __bf16 v16bf;
typedef __attribute__((ext_vector_type(8)))  float  v8f;
typedef __attribute__((ext_vector_type(4)))  int    v4i;

// pointer-to-global-v4i and pointer-to-LDS-v4i
typedef __attribute__((address_space(1))) v4i* gptr_v4i;
typedef __attribute__((address_space(3))) v4i* lptr_v4i;

#define SCALING_F 2.0f   // lora_alpha / lora_rank

union FragU { uint4 u[2]; v16bf v; };

// ---------------------------------------------------------------------------
// CDNA5 async global->LDS staging (ASYNCcnt path), with safe fallback
// ---------------------------------------------------------------------------
#if defined(__gfx1250__) && \
    __has_builtin(__builtin_amdgcn_global_load_async_to_lds_b128) && \
    __has_builtin(__builtin_amdgcn_s_wait_asynccnt)
#define USE_ASYNC_LDS 1
#else
#define USE_ASYNC_LDS 0
#endif

__device__ __forceinline__ void stage16(const __bf16* g, __bf16* l) {
#if USE_ASYNC_LDS
    __builtin_amdgcn_global_load_async_to_lds_b128(
        (gptr_v4i)g, (lptr_v4i)l, 0, 0);
#else
    *(uint4*)l = *(const uint4*)g;
#endif
}

__device__ __forceinline__ void stage_wait() {
#if USE_ASYNC_LDS
    __builtin_amdgcn_s_wait_asynccnt(0);
#endif
}

// ---------------------------------------------------------------------------
// Cast fp32 -> bf16 (grid-stride)
// ---------------------------------------------------------------------------
__global__ void cast_bf16_kernel(const float* __restrict__ in,
                                 __bf16* __restrict__ out, int n) {
    for (int i = blockIdx.x * blockDim.x + threadIdx.x; i < n;
         i += gridDim.x * blockDim.x)
        out[i] = (__bf16)in[i];
}

// ---------------------------------------------------------------------------
// Cast + transpose 1024x1024 weight: wt[n][k] = (bf16) w[k][n]
// block 256 = (k:64, n:4); grid (16, 256)
// ---------------------------------------------------------------------------
__global__ void castT_kernel(const float* __restrict__ w,
                             __bf16* __restrict__ wt) {
    int k = blockIdx.x * 64 + (threadIdx.x & 63);
    int n = blockIdx.y * 4 + (threadIdx.x >> 6);
    wt[(size_t)n * 1024 + k] = (__bf16)w[(size_t)k * 1024 + n];
}

// ---------------------------------------------------------------------------
// LoRA down-projection: xa[row][r] = sum_k x[row][k] * a[k][r]   (r = 8)
// ---------------------------------------------------------------------------
__global__ void lora_down_kernel(const float* __restrict__ x,
                                 const float* __restrict__ a,
                                 float* __restrict__ xa) {
    int idx = blockIdx.x * blockDim.x + threadIdx.x;
    int row = idx >> 3, r = idx & 7;
    float s = 0.f;
    for (int k = 0; k < 1024; ++k)
        s += x[(size_t)row * 1024 + k] * a[k * 8 + r];
    xa[row * 8 + r] = s;
}

// ---------------------------------------------------------------------------
// bf16 GEMM  Y[8192 x 1024] = A[8192 x 1024] @ W  (Wt given transposed)
// epilogue: + bias + SCALING * (xa @ u)      (xa==nullptr disables LoRA)
// store modes: 1 = bf16 [B,H,S,d]   2 = bf16 [B,H,d,S]   3 = f32 row-major
// block 128x128, 128 threads = 4 waves; each wave 32 rows x 128 cols.
// Double-buffered LDS: async DMA of tile k+1 overlaps WMMAs of tile k.
// ---------------------------------------------------------------------------
__global__ __launch_bounds__(128) void gemm_bf16_kernel(
    const __bf16* __restrict__ A,    // [8192 x 1024] row-major
    const __bf16* __restrict__ Wt,   // [1024 x 1024] row-major (Wt[n][k])
    const float*  __restrict__ bias, // [1024]
    const float*  __restrict__ xa,   // [8192 x 8] or null
    const float*  __restrict__ u,    // [8 x 1024] or null
    __bf16* __restrict__ outb,       // bf16 out (modes 1/2)
    float*  __restrict__ outf,       // f32 out (mode 3)
    int mode)
{
    __shared__ __align__(16) __bf16 As[2 * 128 * 32];
    __shared__ __align__(16) __bf16 Bs[2 * 128 * 32];

    const int t    = threadIdx.x;
    const int lane = t & 31;
    const int w    = t >> 5;        // wave id 0..3
    const int hl   = lane >> 4;     // half (0/1)
    const int ln   = lane & 15;

    const int bm = blockIdx.x;      // 0..63
    const int bn = blockIdx.y;      // 0..7

    v8f acc[2][8];
#pragma unroll
    for (int r2 = 0; r2 < 2; ++r2)
#pragma unroll
        for (int i = 0; i < 8; ++i)
#pragma unroll
            for (int j = 0; j < 8; ++j) acc[r2][i][j] = 0.f;

    // stage A/Bt tile pair (8KB each) into buffer `buf` for k-offset k0
    auto stage_tiles = [&](int buf, int k0) {
#pragma unroll
        for (int i = 0; i < 4; ++i) {
            const int c   = t + 128 * i;      // 0..511 (16B chunks)
            const int row = c >> 2;           // 0..127
            const int off = (c & 3) * 8;      // element offset
            stage16(A  + (size_t)(bm * 128 + row) * 1024 + k0 + off,
                    As + buf * 4096 + row * 32 + off);
            stage16(Wt + (size_t)(bn * 128 + row) * 1024 + k0 + off,
                    Bs + buf * 4096 + row * 32 + off);
        }
    };

    stage_tiles(0, 0);
    stage_wait();
    __syncthreads();

    for (int kk = 0; kk < 32; ++kk) {
        const int cur = kk & 1;
        if (kk + 1 < 32) stage_tiles(cur ^ 1, (kk + 1) * 32);  // overlap DMA

        const __bf16* Ac = As + cur * 4096;
        const __bf16* Bc = Bs + cur * 4096;

        FragU a0, a1;
        {   // A fragments: rows w*32+ln and w*32+16+ln (ISA A layout)
            const int e0 = (w * 32 + ln) * 32 + hl * 8;
            a0.u[0] = *(const uint4*)(Ac + e0);
            a0.u[1] = *(const uint4*)(Ac + e0 + 16);
            const int e1 = (w * 32 + 16 + ln) * 32 + hl * 8;
            a1.u[0] = *(const uint4*)(Ac + e1);
            a1.u[1] = *(const uint4*)(Ac + e1 + 16);
        }
#pragma unroll
        for (int nt = 0; nt < 8; ++nt) {
            FragU b;   // Bt fragment: row n, 8 elems at hl*8 and 8 elems at 16+hl*8
            const int f0 = (nt * 16 + ln) * 32 + hl * 8;
            b.u[0] = *(const uint4*)(Bc + f0);
            b.u[1] = *(const uint4*)(Bc + f0 + 16);
            acc[0][nt] = __builtin_amdgcn_wmma_f32_16x16x32_bf16(
                false, a0.v, false, b.v, (short)0, acc[0][nt], false, false);
            acc[1][nt] = __builtin_amdgcn_wmma_f32_16x16x32_bf16(
                false, a1.v, false, b.v, (short)0, acc[1][nt], false, false);
        }

        stage_wait();      // next tile DMA done (own wave)
        __syncthreads();   // all waves: done reading cur, next buffer ready
    }

    // epilogue
#pragma unroll
    for (int r2 = 0; r2 < 2; ++r2) {
#pragma unroll
        for (int nt = 0; nt < 8; ++nt) {
            const int col = bn * 128 + nt * 16 + ln;
            const float bv = bias[col];
#pragma unroll
            for (int v = 0; v < 8; ++v) {
                const int row = bm * 128 + w * 32 + r2 * 16 + hl * 8 + v;
                float val = acc[r2][nt][v] + bv;
                if (xa != nullptr) {
                    float lr = 0.f;
#pragma unroll
                    for (int r = 0; r < 8; ++r)
                        lr += xa[row * 8 + r] * u[r * 1024 + col];
                    val += SCALING_F * lr;
                }
                if (mode == 3) {
                    outf[(size_t)row * 1024 + col] = val;
                } else {
                    const __bf16 bb = (__bf16)val;
                    const int b_ = row >> 11, s_ = row & 2047;
                    const int h_ = col >> 6,  d_ = col & 63;
                    if (mode == 1)   // [B,H,S,d]
                        outb[(((size_t)(b_ * 16 + h_) * 2048) + s_) * 64 + d_] = bb;
                    else             // [B,H,d,S]
                        outb[(((size_t)(b_ * 16 + h_) * 64) + d_) * 2048 + s_] = bb;
                }
            }
        }
    }
}

// ---------------------------------------------------------------------------
// Flash attention: one block = 64 query rows of one (b,h); 128 threads = 4 waves
// Q,K: [BH][2048][64] bf16; Vt: [BH][64][2048] bf16; O: [B][2048][1024] bf16
// Double-buffered K/Vt tiles: DMA of tile kt+1 overlaps compute of tile kt.
// ---------------------------------------------------------------------------
__global__ __launch_bounds__(128) void attn_kernel(
    const __bf16* __restrict__ Q,
    const __bf16* __restrict__ K,
    const __bf16* __restrict__ Vt,
    __bf16* __restrict__ O)
{
    __shared__ __align__(16) __bf16 Qs[64 * 64];
    __shared__ __align__(16) __bf16 Ks[2 * 64 * 64];
    __shared__ __align__(16) __bf16 Vs[2 * 64 * 64];   // Vt tile: [d][key]
    __shared__ __align__(16) __bf16 Ps[4][16 * 64];    // per-wave P staging

    const int t    = threadIdx.x;
    const int lane = t & 31;
    const int w    = t >> 5;      // wave 0..3
    const int hl   = lane >> 4;
    const int ln   = lane & 15;

    const int qb = blockIdx.x;    // 0..31 query block
    const int bh = blockIdx.y;    // 0..63

    const __bf16* Qg = Q  + (size_t)bh * 2048 * 64;
    const __bf16* Kg = K  + (size_t)bh * 2048 * 64;
    const __bf16* Vg = Vt + (size_t)bh * 64 * 2048;

    auto stage_kv = [&](int buf, int kt) {
#pragma unroll
        for (int i = 0; i < 4; ++i) {
            const int c   = t + 128 * i;
            const int row = c >> 3;           // 0..63
            const int off = (c & 7) * 8;      // 0..56 elements
            stage16(Kg + (size_t)(kt * 64 + row) * 64 + off,
                    Ks + buf * 4096 + row * 64 + off);
            stage16(Vg + (size_t)row * 2048 + kt * 64 + off,
                    Vs + buf * 4096 + row * 64 + off);
        }
    };

    {   // prologue: stage Q block + first K/V tile in one issue batch
#pragma unroll
        for (int i = 0; i < 4; ++i) {
            const int c   = t + 128 * i;
            const int row = c >> 3;
            const int off = (c & 7) * 8;
            stage16(Qg + (size_t)(qb * 64 + row) * 64 + off,
                    Qs + row * 64 + off);
        }
        stage_kv(0, 0);
        stage_wait();
    }
    __syncthreads();

    FragU aQ[2];
#pragma unroll
    for (int kc = 0; kc < 2; ++kc) {
        const int e0 = (w * 16 + ln) * 64 + kc * 32 + hl * 8;
        aQ[kc].u[0] = *(const uint4*)(Qs + e0);
        aQ[kc].u[1] = *(const uint4*)(Qs + e0 + 16);
    }

    float mrow[8], lrow[8];
    v8f   Oacc[4];
#pragma unroll
    for (int v = 0; v < 8; ++v) { mrow[v] = -1e30f; lrow[v] = 0.f; }
#pragma unroll
    for (int nt = 0; nt < 4; ++nt)
#pragma unroll
        for (int v = 0; v < 8; ++v) Oacc[nt][v] = 0.f;

    for (int kt = 0; kt < 32; ++kt) {
        const int cur = kt & 1;
        if (kt + 1 < 32) stage_kv(cur ^ 1, kt + 1);   // overlap DMA

        const __bf16* Kc = Ks + cur * 4096;
        const __bf16* Vc = Vs + cur * 4096;

        // scores S = Q K^T * 1/sqrt(64)
        v8f sc[4];
#pragma unroll
        for (int nt = 0; nt < 4; ++nt) {
#pragma unroll
            for (int j = 0; j < 8; ++j) sc[nt][j] = 0.f;
#pragma unroll
            for (int kc = 0; kc < 2; ++kc) {
                FragU bK;
                const int f0 = (nt * 16 + ln) * 64 + kc * 32 + hl * 8;
                bK.u[0] = *(const uint4*)(Kc + f0);
                bK.u[1] = *(const uint4*)(Kc + f0 + 16);
                sc[nt] = __builtin_amdgcn_wmma_f32_16x16x32_bf16(
                    false, aQ[kc].v, false, bK.v, (short)0, sc[nt], false, false);
            }
#pragma unroll
            for (int j = 0; j < 8; ++j) sc[nt][j] *= 0.125f;
        }

        // online softmax update
        float tm[8];
#pragma unroll
        for (int v = 0; v < 8; ++v) {
            float m = sc[0][v];
#pragma unroll
            for (int nt = 1; nt < 4; ++nt) m = fmaxf(m, sc[nt][v]);
#pragma unroll
            for (int off = 1; off < 16; off <<= 1)
                m = fmaxf(m, __shfl_xor(m, off, 32));
            tm[v] = m;
        }
        float alpha[8], rsum[8];
#pragma unroll
        for (int v = 0; v < 8; ++v) {
            const float mn = fmaxf(mrow[v], tm[v]);
            alpha[v] = __expf(mrow[v] - mn);
            mrow[v]  = mn;
            rsum[v]  = 0.f;
        }
#pragma unroll
        for (int nt = 0; nt < 4; ++nt) {
#pragma unroll
            for (int v = 0; v < 8; ++v) {
                Oacc[nt][v] *= alpha[v];
                const float pv = __expf(sc[nt][v] - mrow[v]);
                rsum[v] += pv;
                Ps[w][(hl * 8 + v) * 64 + nt * 16 + ln] = (__bf16)pv;
            }
        }
#pragma unroll
        for (int v = 0; v < 8; ++v) {
            float s = rsum[v];
#pragma unroll
            for (int off = 1; off < 16; off <<= 1)
                s += __shfl_xor(s, off, 32);
            lrow[v] = lrow[v] * alpha[v] + s;
        }

        // O += P @ V   (A = P from per-wave LDS, Bt = Vt tile)
        FragU aP[2];
#pragma unroll
        for (int kc = 0; kc < 2; ++kc) {
            const int e0 = ln * 64 + kc * 32 + hl * 8;
            aP[kc].u[0] = *(const uint4*)(&Ps[w][e0]);
            aP[kc].u[1] = *(const uint4*)(&Ps[w][e0 + 16]);
        }
#pragma unroll
        for (int nt = 0; nt < 4; ++nt) {
#pragma unroll
            for (int kc = 0; kc < 2; ++kc) {
                FragU bV;
                const int f0 = (nt * 16 + ln) * 64 + kc * 32 + hl * 8;
                bV.u[0] = *(const uint4*)(Vc + f0);
                bV.u[1] = *(const uint4*)(Vc + f0 + 16);
                Oacc[nt] = __builtin_amdgcn_wmma_f32_16x16x32_bf16(
                    false, aP[kc].v, false, bV.v, (short)0, Oacc[nt], false, false);
            }
        }

        stage_wait();      // next K/V tile DMA done (own wave)
        __syncthreads();   // all waves: done reading cur, next buffer ready
    }

    // normalize + store O[b][s][h*64+d] (bf16)
    const int b_ = bh >> 4, h_ = bh & 15;
#pragma unroll
    for (int v = 0; v < 8; ++v) {
        const float inv = 1.f / lrow[v];
        const int s_ = qb * 64 + w * 16 + hl * 8 + v;
#pragma unroll
        for (int nt = 0; nt < 4; ++nt) {
            const int d_ = nt * 16 + ln;
            O[((size_t)(b_ * 2048 + s_)) * 1024 + h_ * 64 + d_] =
                (__bf16)(Oacc[nt][v] * inv);
        }
    }
}

// ---------------------------------------------------------------------------
// host launch
// ---------------------------------------------------------------------------
extern "C" void kernel_launch(void* const* d_in, const int* in_sizes, int n_in,
                              void* d_out, int out_size, void* d_ws, size_t ws_size,
                              hipStream_t stream) {
    const float* x   = (const float*)d_in[0];
    const float* w_q = (const float*)d_in[1];
    const float* b_q = (const float*)d_in[2];
    const float* w_k = (const float*)d_in[3];
    const float* b_k = (const float*)d_in[4];
    const float* w_v = (const float*)d_in[5];
    const float* b_v = (const float*)d_in[6];
    const float* w_o = (const float*)d_in[7];
    const float* b_o = (const float*)d_in[8];
    const float* a_q = (const float*)d_in[9];
    const float* u_q = (const float*)d_in[10];
    const float* a_k = (const float*)d_in[11];
    const float* u_k = (const float*)d_in[12];
    const float* a_v = (const float*)d_in[13];
    const float* u_v = (const float*)d_in[14];

    char* p = (char*)d_ws;
    auto take = [&](size_t b) { char* r = p; p += (b + 255) & ~(size_t)255; return r; };

    __bf16* xb  = (__bf16*)take(8192ull * 1024 * 2);
    __bf16* wqt = (__bf16*)take(1024ull * 1024 * 2);
    __bf16* wkt = (__bf16*)take(1024ull * 1024 * 2);
    __bf16* wvt = (__bf16*)take(1024ull * 1024 * 2);
    __bf16* wot = (__bf16*)take(1024ull * 1024 * 2);
    __bf16* qb  = (__bf16*)take(8192ull * 1024 * 2);   // [B,H,S,d]
    __bf16* kb  = (__bf16*)take(8192ull * 1024 * 2);   // [B,H,S,d]
    __bf16* vtb = (__bf16*)take(8192ull * 1024 * 2);   // [B,H,d,S]
    __bf16* ob  = (__bf16*)take(8192ull * 1024 * 2);   // [B,S,H*d]
    float*  xaq = (float*)take(8192ull * 8 * 4);
    float*  xak = (float*)take(8192ull * 8 * 4);
    float*  xav = (float*)take(8192ull * 8 * 4);

    // 1) casts
    cast_bf16_kernel<<<2048, 256, 0, stream>>>(x, xb, 8192 * 1024);
    castT_kernel<<<dim3(16, 256), 256, 0, stream>>>(w_q, wqt);
    castT_kernel<<<dim3(16, 256), 256, 0, stream>>>(w_k, wkt);
    castT_kernel<<<dim3(16, 256), 256, 0, stream>>>(w_v, wvt);
    castT_kernel<<<dim3(16, 256), 256, 0, stream>>>(w_o, wot);

    // 2) LoRA down-projections (f32, rank 8)
    lora_down_kernel<<<256, 256, 0, stream>>>(x, a_q, xaq);
    lora_down_kernel<<<256, 256, 0, stream>>>(x, a_k, xak);
    lora_down_kernel<<<256, 256, 0, stream>>>(x, a_v, xav);

    // 3) fused projections (bf16 WMMA GEMM + bias + LoRA epilogue)
    gemm_bf16_kernel<<<dim3(64, 8), 128, 0, stream>>>(
        xb, wqt, b_q, xaq, u_q, qb, nullptr, 1);
    gemm_bf16_kernel<<<dim3(64, 8), 128, 0, stream>>>(
        xb, wkt, b_k, xak, u_k, kb, nullptr, 1);
    gemm_bf16_kernel<<<dim3(64, 8), 128, 0, stream>>>(
        xb, wvt, b_v, xav, u_v, vtb, nullptr, 2);

    // 4) flash attention
    attn_kernel<<<dim3(32, 64), 128, 0, stream>>>(qb, kb, vtb, ob);

    // 5) output projection -> f32 d_out
    gemm_bf16_kernel<<<dim3(64, 8), 128, 0, stream>>>(
        ob, wot, b_o, nullptr, nullptr, nullptr, (float*)d_out, 3);
}
